// JanusVisionAttention_13331578486996
// MI455X (gfx1250) — compile-verified, matrix operand order checked by reading, not run on hardware
//
#include <hip/hip_runtime.h>
#include <hip/hip_bf16.h>

// ---------------------------------------------------------------------------
// JanusVisionAttention for MI455X (gfx1250, wave32, WMMA bf16 path)
// B=8, S=1024, E=1024, H=16, D=64
// Uses: v_wmma_f32_16x16x32_bf16, global_load_async_to_lds_b128 (ASYNCcnt)
// with double-buffered LDS tiles in both GEMM and flash attention, and
// DPP16 butterfly reductions (no LDS traffic) for the online softmax.
// ---------------------------------------------------------------------------

typedef __bf16 bf16_t;
typedef __attribute__((ext_vector_type(4)))  __bf16 v4bf;
typedef __attribute__((ext_vector_type(8)))  __bf16 v8bf;
typedef __attribute__((ext_vector_type(16))) __bf16 v16bf;
typedef __attribute__((ext_vector_type(8)))  float  v8f;

static __device__ __forceinline__ v16bf mk16(v8bf lo, v8bf hi) {
  v16bf r;
#pragma unroll
  for (int i = 0; i < 8; ++i) { r[i] = lo[i]; r[i + 8] = hi[i]; }
  return r;
}

static __device__ __forceinline__ v8f wmma_bf16(v16bf a, v16bf b, v8f c) {
  // D(16x16 f32) = A(16x32 bf16) x B(32x16 bf16) + C
  return __builtin_amdgcn_wmma_f32_16x16x32_bf16(
      /*neg_a=*/false, a, /*neg_b=*/false, b,
      /*c_mod=*/(short)0, c, /*reuse_a=*/false, /*reuse_b=*/false);
}

// Generic LDS pointer low 32 bits == LDS byte address (CDNA5 aperture rule).
static __device__ __forceinline__ unsigned lds_addr_of(const void* p) {
  return (unsigned)(unsigned long long)p;
}

// CDNA5 async copy global -> LDS, 16 bytes per lane, tracked by ASYNCcnt.
static __device__ __forceinline__ void async_copy_b128(unsigned lds_addr,
                                                       const void* gptr) {
  asm volatile("global_load_async_to_lds_b128 %0, %1, off"
               :: "v"(lds_addr), "v"(gptr)
               : "memory");
}

static __device__ __forceinline__ void wait_async0() {
  asm volatile("s_wait_asynccnt 0" ::: "memory");
}

// ---- DPP16 butterfly reductions within each 16-lane row (VALU only) -------
// quad_perm(1,0,3,2)=0xB1, quad_perm(2,3,0,1)=0x4E, row_ror:4=0x124,
// row_ror:8=0x128. Sequence combines all 16 lanes (rotation butterfly).
template <int CTRL>
static __device__ __forceinline__ float dpp_mov_f32(float x) {
  int y = __builtin_amdgcn_update_dpp(0, __float_as_int(x), CTRL, 0xf, 0xf,
                                      true);
  return __int_as_float(y);
}
static __device__ __forceinline__ float rowmax16(float x) {
  x = fmaxf(x, dpp_mov_f32<0xB1>(x));
  x = fmaxf(x, dpp_mov_f32<0x4E>(x));
  x = fmaxf(x, dpp_mov_f32<0x124>(x));
  x = fmaxf(x, dpp_mov_f32<0x128>(x));
  return x;
}
static __device__ __forceinline__ float rowsum16(float x) {
  x += dpp_mov_f32<0xB1>(x);
  x += dpp_mov_f32<0x4E>(x);
  x += dpp_mov_f32<0x124>(x);
  x += dpp_mov_f32<0x128>(x);
  return x;
}

// ---------------------------------------------------------------------------
// Kernel 0: fp32 -> bf16 cast (vectorized x4)
// ---------------------------------------------------------------------------
__global__ void jva_cast_bf16(const float* __restrict__ in,
                              bf16_t* __restrict__ out, int n4) {
  int i = blockIdx.x * blockDim.x + threadIdx.x;
  if (i < n4) {
    float4 v = ((const float4*)in)[i];
    v4bf o = { (bf16_t)v.x, (bf16_t)v.y, (bf16_t)v.z, (bf16_t)v.w };
    ((v4bf*)out)[i] = o;
  }
}

// ---------------------------------------------------------------------------
// Kernel 1: C[M,N] (f32) = A[M,K](bf16) * B[N,K](bf16)^T + bias[N]
// Block tile 128x128, BK=32, 8 wave32s (4x2), each wave 32x64 (2x4 WMMA).
// Double-buffered LDS fed by global_load_async_to_lds_b128.
// ---------------------------------------------------------------------------
#define GBM 128
#define GBN 128
#define GBK 32
#define GPAD 8   // 40 elems = 80B = 5*16B: keeps b128 alignment, staggers banks

template <int N, int K>
__global__ __launch_bounds__(256, 2)
void jva_gemm_bf16(const bf16_t* __restrict__ A,   // [M,K]
                   const bf16_t* __restrict__ B,   // [N,K]
                   const float*  __restrict__ bias,// [N]
                   float* __restrict__ C) {        // [M,N]
  constexpr int LDT = GBK + GPAD;
  __shared__ bf16_t As[2][GBM][LDT];
  __shared__ bf16_t Bs[2][GBN][LDT];

  const int tid  = threadIdx.x;
  const int wid  = tid >> 5;
  const int lane = tid & 31;
  const int half = lane >> 4;   // which 16-lane half
  const int l16  = lane & 15;

  const int mBlock = blockIdx.y * GBM;
  const int nBlock = blockIdx.x * GBN;
  const int wm = (wid & 3) * 32;   // wave rows: 2 x 16
  const int wn = (wid >> 2) * 64;  // wave cols: 4 x 16

  // Per-thread async copy assignment: 2 A chunks + 2 B chunks of 16B.
  const int r0 = tid >> 2;            // rows 0..63
  const int c0 = (tid & 3) * 8;       // col 0/8/16/24
  auto issue_tile = [&](int buf, int k0) {
#pragma unroll
    for (int it = 0; it < 2; ++it) {
      int row = r0 + it * 64;
      async_copy_b128(lds_addr_of(&As[buf][row][c0]),
                      &A[(size_t)(mBlock + row) * K + k0 + c0]);
      async_copy_b128(lds_addr_of(&Bs[buf][row][c0]),
                      &B[(size_t)(nBlock + row) * K + k0 + c0]);
    }
  };

  v8f acc[2][4];
#pragma unroll
  for (int i = 0; i < 2; ++i)
#pragma unroll
    for (int j = 0; j < 4; ++j) acc[i][j] = {};

  issue_tile(0, 0);
  wait_async0();
  __syncthreads();

  for (int k0 = 0; k0 < K; k0 += GBK) {
    const int cur = (k0 / GBK) & 1;
    if (k0 + GBK < K) issue_tile(cur ^ 1, k0 + GBK);  // overlap with WMMAs

    // Fragment loads per documented 16-bit layout:
    // lane<16: K{0..7, 16..23}; lane>=16: K{8..15, 24..31}
    v16bf af[2], bf[4];
#pragma unroll
    for (int i = 0; i < 2; ++i) {
      const bf16_t* ar = &As[cur][wm + i * 16 + l16][0];
      af[i] = mk16(*(const v8bf*)(ar + half * 8),
                   *(const v8bf*)(ar + 16 + half * 8));
    }
#pragma unroll
    for (int j = 0; j < 4; ++j) {
      const bf16_t* br = &Bs[cur][wn + j * 16 + l16][0];
      bf[j] = mk16(*(const v8bf*)(br + half * 8),
                   *(const v8bf*)(br + 16 + half * 8));
    }
#pragma unroll
    for (int i = 0; i < 2; ++i)
#pragma unroll
      for (int j = 0; j < 4; ++j)
        acc[i][j] = wmma_bf16(af[i], bf[j], acc[i][j]);

    wait_async0();     // next tile landed (this thread's copies)
    __syncthreads();   // all waves' copies landed / done reading cur
  }

  // Epilogue: C/D layout: VGPR r -> M = r + half*8, N = lane&15.
  // N is constexpr -> immediate store offsets.
  float* cbase = C + (size_t)(mBlock + wm + half * 8) * N + nBlock + wn + l16;
#pragma unroll
  for (int j = 0; j < 4; ++j) {
    float bv = bias[nBlock + wn + j * 16 + l16];
#pragma unroll
    for (int i = 0; i < 2; ++i)
#pragma unroll
      for (int r = 0; r < 8; ++r)
        cbase[(size_t)(i * 16 + r) * N + j * 16] = acc[i][j][r] + bv;
  }
}

// ---------------------------------------------------------------------------
// Kernel 2: QK LayerNorm + repack. One wave32 per (b,s,h), 2 elems/lane.
// qkv f32 [B,S,3,H,D] -> Q,K bf16 [B,H,S,D], V^T bf16 [B,H,D,S]
// ---------------------------------------------------------------------------
__global__ __launch_bounds__(256)
void jva_ln_pack(const float* __restrict__ qkv,
                 const float* __restrict__ qg, const float* __restrict__ qb,
                 const float* __restrict__ kg, const float* __restrict__ kb,
                 bf16_t* __restrict__ Qo, bf16_t* __restrict__ Ko,
                 bf16_t* __restrict__ Vt) {
  constexpr int S = 1024, H = 16, Dh = 64, E = 1024;
  const int gw   = (blockIdx.x * blockDim.x + threadIdx.x) >> 5;
  const int lane = threadIdx.x & 31;
  const int h = gw % H;
  const int s = (gw / H) % S;
  const int b = gw / (H * S);

  const float* base = qkv + ((size_t)b * S + s) * (3 * E) + h * Dh;
  float q0 = base[lane],         q1 = base[lane + 32];
  float k0 = base[E + lane],     k1 = base[E + lane + 32];
  float v0 = base[2 * E + lane], v1 = base[2 * E + lane + 32];

  float sq = q0 + q1, sq2 = q0 * q0 + q1 * q1;
  float sk = k0 + k1, sk2 = k0 * k0 + k1 * k1;
#pragma unroll
  for (int off = 16; off >= 1; off >>= 1) {
    sq  += __shfl_xor(sq,  off, 32);
    sq2 += __shfl_xor(sq2, off, 32);
    sk  += __shfl_xor(sk,  off, 32);
    sk2 += __shfl_xor(sk2, off, 32);
  }
  const float inv = 1.0f / 64.0f;
  float qmu = sq * inv, qvar = sq2 * inv - qmu * qmu;
  float kmu = sk * inv, kvar = sk2 * inv - kmu * kmu;
  float qr = rsqrtf(qvar + 1e-5f);
  float kr = rsqrtf(kvar + 1e-5f);
  const float att_scale = 0.125f;  // D^-0.5 = 64^-0.5

  float oq0 = ((q0 - qmu) * qr * qg[lane]      + qb[lane])      * att_scale;
  float oq1 = ((q1 - qmu) * qr * qg[lane + 32] + qb[lane + 32]) * att_scale;
  float ok0 =  (k0 - kmu) * kr * kg[lane]      + kb[lane];
  float ok1 =  (k1 - kmu) * kr * kg[lane + 32] + kb[lane + 32];

  const size_t bh = (size_t)b * H + h;
  bf16_t* qdst = Qo + (bh * S + s) * Dh;
  bf16_t* kdst = Ko + (bh * S + s) * Dh;
  qdst[lane] = (bf16_t)oq0;  qdst[lane + 32] = (bf16_t)oq1;
  kdst[lane] = (bf16_t)ok0;  kdst[lane + 32] = (bf16_t)ok1;
  Vt[(bh * Dh + lane) * S + s]      = (bf16_t)v0;
  Vt[(bh * Dh + lane + 32) * S + s] = (bf16_t)v1;
}

// ---------------------------------------------------------------------------
// Kernel 3: Flash attention. Grid: (S/128, B*H), 256 threads (8 wave32).
// Each wave: 16 queries. K/V^T streamed in double-buffered 64-key tiles via
// async-to-LDS; online softmax uses DPP16 reductions (no LDS traffic).
// Q bf16 [B,H,S,D], K bf16 [B,H,S,D], V^T bf16 [B,H,D,S]
// -> O bf16 [B,S,E] (token-major, ready for output projection)
// ---------------------------------------------------------------------------
#define FKT 64
#define FPAD 8   // 72 elems = 144B = 9*16B: aligned, bank-staggered

__global__ __launch_bounds__(256, 2)
void jva_flash_attn(const bf16_t* __restrict__ Q,
                    const bf16_t* __restrict__ Kx,
                    const bf16_t* __restrict__ Vt,
                    bf16_t* __restrict__ O) {
  constexpr int S = 1024, Dh = 64, H = 16, E = 1024;
  constexpr int NT = S / FKT;                    // 16 key tiles
  __shared__ bf16_t Ks[2][FKT][Dh + FPAD];       // [key][d]
  __shared__ bf16_t Vs[2][Dh][FKT + FPAD];       // [d][key]  (V^T tile)
  __shared__ bf16_t Ps[8][16][FKT + FPAD];       // per-wave P staging [q][key]

  const int tid  = threadIdx.x;
  const int wid  = tid >> 5;
  const int lane = tid & 31;
  const int half = lane >> 4;
  const int l16  = lane & 15;

  const int bh = blockIdx.y;
  const int b  = bh / H;
  const int h  = bh % H;
  const int qBase = blockIdx.x * 128 + wid * 16;

  const bf16_t* Qp = Q  + ((size_t)bh * S + qBase) * Dh;
  const bf16_t* Kp = Kx + (size_t)bh * S * Dh;
  const bf16_t* Vp = Vt + (size_t)bh * Dh * S;

  // Q fragments (held in registers for whole kernel): 2 K-steps of D=64
  v16bf qa[2];
#pragma unroll
  for (int ks = 0; ks < 2; ++ks) {
    const bf16_t* qrow = Qp + (size_t)l16 * Dh + ks * 32;
    qa[ks] = mk16(*(const v8bf*)(qrow + half * 8),
                  *(const v8bf*)(qrow + 16 + half * 8));
  }

  float m[8], l[8];
  v8f o[4];
#pragma unroll
  for (int r = 0; r < 8; ++r) { m[r] = -__builtin_inff(); l[r] = 0.0f; }
#pragma unroll
  for (int j = 0; j < 4; ++j) o[j] = {};

  const int lr = tid >> 3;            // 0..31: tile row
  const int lc = (tid & 7) * 8;       // 16B chunk within 64-wide row

  auto issue_tile = [&](int buf, int k0) {
#pragma unroll
    for (int it = 0; it < 2; ++it) {
      int row = lr + it * 32;
      async_copy_b128(lds_addr_of(&Ks[buf][row][lc]),
                      &Kp[(size_t)(k0 + row) * Dh + lc]);
      async_copy_b128(lds_addr_of(&Vs[buf][row][lc]),
                      &Vp[(size_t)row * S + k0 + lc]);
    }
  };

  issue_tile(0, 0);
  wait_async0();
  __syncthreads();

  for (int kt = 0; kt < NT; ++kt) {
    const int cur = kt & 1;
    if (kt + 1 < NT) issue_tile(cur ^ 1, (kt + 1) * FKT);  // overlap w/ compute

    // Scores: s[j] = q(16x64) . k^T for key sub-tile j (16 keys)
    v8f s[4];
#pragma unroll
    for (int j = 0; j < 4; ++j) {
      s[j] = {};
#pragma unroll
      for (int ks = 0; ks < 2; ++ks) {
        const bf16_t* krow = &Ks[cur][j * 16 + l16][ks * 32];
        v16bf kf = mk16(*(const v8bf*)(krow + half * 8),
                        *(const v8bf*)(krow + 16 + half * 8));
        s[j] = wmma_bf16(qa[ks], kf, s[j]);
      }
    }

    // Online softmax per row (row M = r + half*8, spread over 16 lanes).
    // DPP16 butterflies keep the reductions entirely in VALU.
#pragma unroll
    for (int r = 0; r < 8; ++r) {
      float mx = fmaxf(fmaxf(s[0][r], s[1][r]), fmaxf(s[2][r], s[3][r]));
      mx = rowmax16(mx);
      float mn = fmaxf(m[r], mx);
      float sc = __expf(m[r] - mn);
      m[r] = mn;
      float rs = 0.0f;
#pragma unroll
      for (int j = 0; j < 4; ++j) {
        float p = __expf(s[j][r] - mn);
        s[j][r] = p;
        rs += p;
      }
      rs = rowsum16(rs);
      l[r] = l[r] * sc + rs;
#pragma unroll
      for (int j = 0; j < 4; ++j) o[j][r] *= sc;
      // Stage P (bf16) into per-wave LDS in [q][key] order
#pragma unroll
      for (int j = 0; j < 4; ++j)
        Ps[wid][r + half * 8][j * 16 + l16] = (bf16_t)s[j][r];
    }
    __syncthreads();

    // o += P(16x64) . V(64x64): A from Ps, B from Vs (cols = d)
    v16bf pa[2];
#pragma unroll
    for (int ks = 0; ks < 2; ++ks) {
      const bf16_t* prow = &Ps[wid][l16][ks * 32];
      pa[ks] = mk16(*(const v8bf*)(prow + half * 8),
                    *(const v8bf*)(prow + 16 + half * 8));
    }
#pragma unroll
    for (int j = 0; j < 4; ++j) {
#pragma unroll
      for (int ks = 0; ks < 2; ++ks) {
        const bf16_t* vrow = &Vs[cur][j * 16 + l16][ks * 32];
        v16bf vf = mk16(*(const v8bf*)(vrow + half * 8),
                        *(const v8bf*)(vrow + 16 + half * 8));
        o[j] = wmma_bf16(pa[ks], vf, o[j]);
      }
    }
    wait_async0();     // next K/V tile landed (this thread's copies)
    __syncthreads();   // all copies landed; all waves done with cur
  }

  // Normalize (reciprocal once per row) and write bf16 out in [B,S,E]
  float rl[8];
#pragma unroll
  for (int r = 0; r < 8; ++r) rl[r] = 1.0f / l[r];
#pragma unroll
  for (int j = 0; j < 4; ++j) {
#pragma unroll
    for (int r = 0; r < 8; ++r) {
      int q = qBase + r + half * 8;
      float val = o[j][r] * rl[r];
      O[((size_t)b * S + q) * E + h * Dh + j * 16 + l16] = (bf16_t)val;
    }
  }
}

// ---------------------------------------------------------------------------
// Host-side orchestration
// ---------------------------------------------------------------------------
extern "C" void kernel_launch(void* const* d_in, const int* in_sizes, int n_in,
                              void* d_out, int out_size, void* d_ws,
                              size_t ws_size, hipStream_t stream) {
  (void)in_sizes; (void)n_in; (void)out_size; (void)ws_size;
  constexpr int B = 8, S = 1024, E = 1024, H = 16, Dh = 64;
  constexpr int M = B * S;          // 8192 tokens
  constexpr int N_QKV = 3 * E;      // 3072
  constexpr int K = E;              // 1024

  const float* hs      = (const float*)d_in[0];
  const float* w_qkv   = (const float*)d_in[1];
  const float* b_qkv   = (const float*)d_in[2];
  const float* q_gamma = (const float*)d_in[3];
  const float* q_beta  = (const float*)d_in[4];
  const float* k_gamma = (const float*)d_in[5];
  const float* k_beta  = (const float*)d_in[6];
  const float* w_proj  = (const float*)d_in[7];
  const float* b_proj  = (const float*)d_in[8];
  float* out = (float*)d_out;

  char* ws = (char*)d_ws;
  size_t off = 0;
  auto take = [&](size_t bytes) -> char* {
    char* p = ws + off;
    off += (bytes + 255) & ~(size_t)255;
    return p;
  };
  bf16_t* hs_bf   = (bf16_t*)take((size_t)M * E * 2);        // 16 MB
  bf16_t* wqkv_bf = (bf16_t*)take((size_t)N_QKV * E * 2);    //  6 MB
  bf16_t* wprj_bf = (bf16_t*)take((size_t)E * E * 2);        //  2 MB
  float*  qkv_f32 = (float*) take((size_t)M * N_QKV * 4);    // 96 MB
  bf16_t* q_bf    = (bf16_t*)take((size_t)B * H * S * Dh * 2);
  bf16_t* k_bf    = (bf16_t*)take((size_t)B * H * S * Dh * 2);
  bf16_t* vT_bf   = (bf16_t*)take((size_t)B * H * Dh * S * 2);
  bf16_t* attn_bf = (bf16_t*)take((size_t)M * E * 2);        // 16 MB

  // 1) Casts to bf16
  {
    int n4 = (M * E) / 4;
    jva_cast_bf16<<<(n4 + 255) / 256, 256, 0, stream>>>(hs, hs_bf, n4);
  }
  {
    int n4 = (N_QKV * E) / 4;
    jva_cast_bf16<<<(n4 + 255) / 256, 256, 0, stream>>>(w_qkv, wqkv_bf, n4);
  }
  {
    int n4 = (E * E) / 4;
    jva_cast_bf16<<<(n4 + 255) / 256, 256, 0, stream>>>(w_proj, wprj_bf, n4);
  }

  // 2) QKV projection: [M,3E] = hs_bf [M,K] x wqkv_bf [3E,K]^T + b_qkv
  {
    dim3 grid(N_QKV / GBN, M / GBM);
    jva_gemm_bf16<N_QKV, K><<<grid, 256, 0, stream>>>(hs_bf, wqkv_bf, b_qkv,
                                                      qkv_f32);
  }

  // 3) QK LayerNorm + repack (one wave32 per (b,s,h))
  {
    int waves = B * S * H;                 // 131072
    jva_ln_pack<<<waves / 8, 256, 0, stream>>>(qkv_f32, q_gamma, q_beta,
                                               k_gamma, k_beta, q_bf, k_bf,
                                               vT_bf);
  }

  // 4) Flash attention
  {
    dim3 grid(S / 128, B * H);
    jva_flash_attn<<<grid, 256, 0, stream>>>(q_bf, k_bf, vT_bf, attn_bf);
  }

  // 5) Output projection: out [M,E] = attn_bf [M,K] x wprj_bf [E,K]^T + b_proj
  {
    dim3 grid(E / GBN, M / GBM);
    jva_gemm_bf16<E, K><<<grid, 256, 0, stream>>>(attn_bf, wprj_bf, b_proj,
                                                  out);
  }
}
